// SPR_RGCN_88648124990022
// MI455X (gfx1250) — matrix operand order repaired and checked by reading, not compiled, not run on hardware
//
#include <hip/hip_runtime.h>

// ---------------- problem constants ----------------
#define NN      50000
#define EE      640000
#define RR      3
#define EMB     64
#define HID     128
#define NCLS    4
#define GG      512
#define IN_DIM  129          // 2*EMB + 1
#define SEGW1   160          // IN_DIM padded to multiple of 32
#define KT1     20           // (4*SEGW1)/32
#define SEGW2   128
#define KT2     16           // (4*SEGW2)/32
#define NT      3125         // NN/16 row tiles (exact)
#define NBINS   (RR * NN)    // 150000 (relation, dst) segments

typedef __attribute__((ext_vector_type(16))) __bf16 v16bf;
typedef __attribute__((ext_vector_type(8)))  float  v8f;

__device__ __forceinline__ unsigned short f2bf(float f) {
    unsigned u = __float_as_uint(f);
    unsigned r = u + 0x7FFFu + ((u >> 16) & 1u);   // round-to-nearest-even
    return (unsigned short)(r >> 16);
}

// WMMA A-fragment swizzle (ISA 7.12.2, 16-bit A 16x32):
// frag layout [row_tile][kt][lane(32)][16 elems], elem e=2v+p holds
// K = (v>=4?16:0) + (v&3)*2 + (lane>>4)*8 + p, row = row_tile*16 + (lane&15)
__device__ __forceinline__ long long afrag_index(int n, int ktot, int nkt) {
    int tile = n >> 4;
    int kt   = ktot >> 5;
    int k32  = ktot & 31;
    int w16  = k32 & 15;
    int half = w16 >> 3;
    int rem  = w16 & 7;
    int v    = ((k32 >= 16) ? 4 : 0) + (rem >> 1);
    int p    = rem & 1;
    int lane = (half << 4) + (n & 15);
    return (((long long)tile * nkt + kt) * 32 + lane) * 16 + (v << 1) + p;
}

// ---------------- zero ----------------
__global__ void zero_f32(float* __restrict__ p, long long n) {
    long long i = (long long)blockIdx.x * blockDim.x + threadIdx.x;
    long long s = (long long)gridDim.x * blockDim.x;
    for (; i < n; i += s) p[i] = 0.0f;
}

// ---------------- build node features x[N,129] ----------------
__global__ void build_x(const float* __restrict__ pos,
                        const int* __restrict__ sid, const int* __restrict__ cid,
                        const float* __restrict__ semb, const float* __restrict__ cemb,
                        float* __restrict__ x) {
    long long idx = (long long)blockIdx.x * blockDim.x + threadIdx.x;
    if (idx >= (long long)NN * IN_DIM) return;
    int n = (int)(idx / IN_DIM);
    int k = (int)(idx % IN_DIM);
    float v;
    if (k < EMB)            v = semb[sid[n] * EMB + k];
    else if (k < 2 * EMB)   v = cemb[cid[n] * EMB + (k - EMB)];
    else                    v = pos[n];
    x[idx] = v;
}

// ---------------- CSR build: histogram -> scan -> fill ----------------
__global__ void hist_edges(const int* __restrict__ ei, const int* __restrict__ et,
                           int* __restrict__ hist) {
    int e = blockIdx.x * blockDim.x + threadIdx.x;
    if (e >= EE) return;
    atomicAdd(&hist[et[e] * NN + ei[EE + e]], 1);
}

__global__ __launch_bounds__(1024) void scan_bins(const int* __restrict__ hist,
                                                  int* __restrict__ offs,
                                                  int* __restrict__ cursor) {
    __shared__ int ts[1024];
    const int t = threadIdx.x;
    const int chunk = (NBINS + 1023) / 1024;       // 147
    int b0 = t * chunk;
    int b1 = b0 + chunk; if (b1 > NBINS) b1 = NBINS;
    int s = 0;
    for (int b = b0; b < b1; ++b) s += hist[b];
    ts[t] = s;
    __syncthreads();
    for (int off = 1; off < 1024; off <<= 1) {     // Hillis-Steele inclusive scan
        int v = ts[t];
        int add = (t >= off) ? ts[t - off] : 0;
        __syncthreads();
        ts[t] = v + add;
        __syncthreads();
    }
    int run = (t == 0) ? 0 : ts[t - 1];
    for (int b = b0; b < b1; ++b) {
        int h = hist[b];
        offs[b] = run;
        cursor[b] = run;
        run += h;
    }
    if (t == 1023) offs[NBINS] = ts[1023];         // == EE
}

__global__ void fill_csr(const int* __restrict__ ei, const int* __restrict__ et,
                         int* __restrict__ cursor, int* __restrict__ srcs) {
    int e = blockIdx.x * blockDim.x + threadIdx.x;
    if (e >= EE) return;
    int bin = et[e] * NN + ei[EE + e];
    int slot = atomicAdd(&cursor[bin], 1);
    srcs[slot] = ei[e];
}

// -------- per-(relation,dst) mean gather, written straight into A frags -------
__global__ void segment_mean_frag(const int* __restrict__ offs, const int* __restrict__ srcs,
                                  const float* __restrict__ feat,
                                  unsigned short* __restrict__ A,
                                  int nkt, int segw, int d) {
    int b    = (int)(((long long)blockIdx.x * blockDim.x + threadIdx.x) >> 5);
    int lane = threadIdx.x & 31;
    if (b >= NBINS) return;
    int beg = offs[b], end = offs[b + 1];
    float acc[5] = {0.f, 0.f, 0.f, 0.f, 0.f};
    for (int j = beg; j < end; ++j) {
        const float* fs = feat + (long long)srcs[j] * d;
#pragma unroll
        for (int q = 0; q < 5; ++q) {
            int i = lane + (q << 5);
            if (i < d) acc[q] += fs[i];
        }
    }
    float c = (float)(end - beg);
    float inv = (c > 0.f) ? 1.0f / c : 0.0f;       // empty bin -> mean 0
    int r = b / NN;
    int n = b % NN;
    int kbase = r * segw;
#pragma unroll
    for (int q = 0; q < 5; ++q) {
        int i = lane + (q << 5);
        if (i < d) A[afrag_index(n, kbase + i, nkt)] = f2bf(acc[q] * inv);
    }
}

// -------- copy layer input (root-transform operand) into segment-3 frags ------
__global__ void feat_to_frag(const float* __restrict__ feat, unsigned short* __restrict__ A,
                             int nkt, int segw, int d) {
    long long idx = (long long)blockIdx.x * blockDim.x + threadIdx.x;
    if (idx >= (long long)NN * d) return;
    int n  = (int)(idx / d);
    int kl = (int)(idx % d);
    A[afrag_index(n, RR * segw + kl, nkt)] = f2bf(feat[idx]);
}

// ---------------- weight prep -> B fragments (bf16, WMMA-swizzled) ------------
// B-frag layout [kt][col_tile(8)][lane(32)][16]; elem e=2v+p holds
// K = (v>=4?16:0) + (lane>>4)*8 + (v&3)*2 + p ; column = col_tile*16 + (lane&15)
__global__ void prep_B(const float* __restrict__ W, const float* __restrict__ root,
                       unsigned short* __restrict__ Bf,
                       int segw, int d, long long total) {
    long long idx = (long long)blockIdx.x * blockDim.x + threadIdx.x;
    if (idx >= total) return;
    int within = (int)(idx & 511);
    long long t = idx >> 9;                        // kt*8 + ct
    int ct = (int)(t & 7);
    int kt = (int)(t >> 3);
    int lane = within >> 4;
    int e    = within & 15;
    int v    = e >> 1, p = e & 1;
    int n    = (ct << 4) + (lane & 15);
    int half = lane >> 4;
    int k32  = ((v >= 4) ? 16 : 0) + (half << 3) + ((v & 3) << 1) + p;
    int ktot = kt * 32 + k32;
    int seg  = ktot / segw;
    int kl   = ktot % segw;
    float val = 0.0f;
    if (kl < d)
        val = (seg < RR) ? W[((long long)seg * d + kl) * HID + n]
                         : root[(long long)kl * HID + n];
    Bf[idx] = f2bf(val);
}

// ---------------- fragment GEMM: C[N,128] = A @ B (+bias, ReLU) ---------------
// 256 threads = 8 wave32s; wave w owns row tile (blockIdx.x*8+w): 16 rows x 128 cols.
// B K-tile double-buffered in LDS; next tile staged in VGPRs across the WMMA
// block so its global-load latency hides behind compute; 8 WMMAs back-to-back.
__global__ __launch_bounds__(256) void gemm_frag(
        const unsigned short* __restrict__ A, const unsigned short* __restrict__ B,
        const float* __restrict__ bias, float* __restrict__ out,
        int nkt, int relu) {
    __shared__ unsigned short Bsh[2][4096];        // 2 x 8KB K-tiles of B
    const int lane = threadIdx.x & 31;
    const int wave = threadIdx.x >> 5;
    const int rt   = blockIdx.x * 8 + wave;
    const bool active = (rt < NT);

    {   // stage K-tile 0
        const uint4* g = (const uint4*)B;
        uint4* l = (uint4*)Bsh[0];
        l[threadIdx.x]       = g[threadIdx.x];
        l[threadIdx.x + 256] = g[threadIdx.x + 256];
    }
    __syncthreads();

    const unsigned short* aptr = A + ((long long)rt * nkt * 32 + lane) * 16;
    v8f acc[8] = {};
    for (int kt = 0; kt < nkt; ++kt) {
        // issue next K-tile's global loads into registers (clamped on last iter)
        int knext = (kt + 1 < nkt) ? (kt + 1) : kt;
        const uint4* g = (const uint4*)(B + (long long)knext * 4096);
        uint4 s0 = g[threadIdx.x];
        uint4 s1 = g[threadIdx.x + 256];
        if (active) {
            __builtin_prefetch(aptr + 512, 0, 3);  // next A fragment, near cache
            v16bf a = *(const v16bf*)aptr;
            const unsigned short* bs = Bsh[kt & 1];
            v16bf bf[8];
#pragma unroll
            for (int c = 0; c < 8; ++c)
                bf[c] = *(const v16bf*)(bs + ((c * 32 + lane) << 4));
#pragma unroll
            for (int c = 0; c < 8; ++c)
                acc[c] = __builtin_amdgcn_wmma_f32_16x16x32_bf16(
                             false, a, false, bf[c], (short)0, acc[c], false, false);
        }
        aptr += 512;
        // store staged tile to the other buffer after compute
        uint4* l = (uint4*)Bsh[(kt + 1) & 1];
        l[threadIdx.x]       = s0;
        l[threadIdx.x + 256] = s1;
        __syncthreads();
    }
    if (active) {
        const int row0 = rt * 16 + ((lane >> 4) << 3);   // + j for VGPR j
        const int nc   = lane & 15;
#pragma unroll
        for (int c = 0; c < 8; ++c) {
            float bv = bias[c * 16 + nc];
#pragma unroll
            for (int j = 0; j < 8; ++j) {
                float v = acc[c][j] + bv;
                if (relu) v = v > 0.0f ? v : 0.0f;
                out[(long long)(row0 + j) * HID + c * 16 + nc] = v;
            }
        }
    }
}

// ---------------- graph segment boundaries (batch is sorted) ------------------
__global__ void graph_bounds(const int* __restrict__ batch, int* __restrict__ gstart) {
    int n = blockIdx.x * blockDim.x + threadIdx.x;
    if (n > NN) return;
    if (n == 0) {
        int b0 = batch[0];
        for (int g = 0; g <= b0; ++g) gstart[g] = 0;
    } else if (n == NN) {
        int bl = batch[NN - 1];
        for (int g = bl + 1; g <= GG; ++g) gstart[g] = NN;
    } else {
        int a = batch[n - 1], b = batch[n];
        for (int g = a + 1; g <= b; ++g) gstart[g] = n;
    }
}

// ---------------- per-graph mean pool: one wave per graph ---------------------
__global__ void pool_mean(const float* __restrict__ h, const int* __restrict__ gstart,
                          float* __restrict__ pooled) {
    int g    = (int)(((long long)blockIdx.x * blockDim.x + threadIdx.x) >> 5);
    int lane = threadIdx.x & 31;
    if (g >= GG) return;
    int beg = gstart[g], end = gstart[g + 1];
    float a0 = 0.f, a1 = 0.f, a2 = 0.f, a3 = 0.f;
    for (int n = beg; n < end; ++n) {
        const float* hs = h + (long long)n * HID;
        a0 += hs[lane];
        a1 += hs[lane + 32];
        a2 += hs[lane + 64];
        a3 += hs[lane + 96];
    }
    float c = (float)(end - beg);
    float inv = (c > 0.f) ? 1.0f / c : 0.0f;
    float* pd = pooled + (long long)g * HID;
    pd[lane]      = a0 * inv;
    pd[lane + 32] = a1 * inv;
    pd[lane + 64] = a2 * inv;
    pd[lane + 96] = a3 * inv;
}

// ---------------- head: out[G,4] = pooled @ lin_w + lin_b ---------------------
__global__ void head(const float* __restrict__ pooled,
                     const float* __restrict__ lw, const float* __restrict__ lb,
                     float* __restrict__ out) {
    int idx = blockIdx.x * blockDim.x + threadIdx.x;
    if (idx >= GG * NCLS) return;
    int g = idx >> 2, c = idx & 3;
    float s = 0.0f;
#pragma unroll 8
    for (int k = 0; k < HID; ++k) s += pooled[g * HID + k] * lw[k * NCLS + c];
    out[idx] = s + lb[c];
}

// ---------------- launcher ----------------
extern "C" void kernel_launch(void* const* d_in, const int* in_sizes, int n_in,
                              void* d_out, int out_size, void* d_ws, size_t ws_size,
                              hipStream_t stream) {
    (void)in_sizes; (void)n_in; (void)out_size; (void)ws_size;
    const float* pos   = (const float*)d_in[0];
    const int*   sid   = (const int*)d_in[1];
    const int*   cid   = (const int*)d_in[2];
    const int*   ei    = (const int*)d_in[3];
    const int*   et    = (const int*)d_in[4];
    const int*   batch = (const int*)d_in[5];
    const float* semb  = (const float*)d_in[6];
    const float* cemb  = (const float*)d_in[7];
    const float* W1    = (const float*)d_in[8];
    const float* root1 = (const float*)d_in[9];
    const float* b1    = (const float*)d_in[10];
    const float* W2    = (const float*)d_in[11];
    const float* root2 = (const float*)d_in[12];
    const float* b2    = (const float*)d_in[13];
    const float* lw    = (const float*)d_in[14];
    const float* lb    = (const float*)d_in[15];
    float* out = (float*)d_out;

    char* base = (char*)d_ws;
    size_t off = 0;
    auto carve = [&](size_t bytes) -> char* {
        char* p = base + off;
        off += (bytes + 255) & ~(size_t)255;
        return p;
    };
    float*          x      = (float*)         carve((size_t)NN * IN_DIM * 4);
    int*            hist   = (int*)           carve((size_t)NBINS * 4);
    int*            offs   = (int*)           carve((size_t)(NBINS + 1) * 4);
    int*            cursor = (int*)           carve((size_t)NBINS * 4);
    int*            srcs   = (int*)           carve((size_t)EE * 4);
    int*            gstart = (int*)           carve((size_t)(GG + 1) * 4);
    unsigned short* Af     = (unsigned short*)carve((size_t)NT * KT1 * 512 * 2);
    unsigned short* B1f    = (unsigned short*)carve((size_t)KT1 * 8 * 512 * 2);
    unsigned short* B2f    = (unsigned short*)carve((size_t)KT2 * 8 * 512 * 2);
    float*          h1     = (float*)         carve((size_t)NN * HID * 4);
    float*          h2     = (float*)         carve((size_t)NN * HID * 4);
    float*          pooled = (float*)         carve((size_t)GG * HID * 4);

    // node features + weight fragments
    hipLaunchKernelGGL(build_x, dim3(((long long)NN * IN_DIM + 255) / 256), dim3(256), 0, stream,
                       pos, sid, cid, semb, cemb, x);
    hipLaunchKernelGGL(prep_B, dim3((KT1 * 8 * 512) / 256), dim3(256), 0, stream,
                       W1, root1, B1f, SEGW1, IN_DIM, (long long)KT1 * 8 * 512);
    hipLaunchKernelGGL(prep_B, dim3((KT2 * 8 * 512) / 256), dim3(256), 0, stream,
                       W2, root2, B2f, SEGW2, HID, (long long)KT2 * 8 * 512);

    // CSR over (relation, dst) segments — built once, reused by both layers
    hipLaunchKernelGGL(zero_f32, dim3(64), dim3(256), 0, stream,
                       (float*)hist, (long long)NBINS);
    hipLaunchKernelGGL(hist_edges, dim3((EE + 255) / 256), dim3(256), 0, stream,
                       ei, et, hist);
    hipLaunchKernelGGL(scan_bins, dim3(1), dim3(1024), 0, stream,
                       hist, offs, cursor);
    hipLaunchKernelGGL(fill_csr, dim3((EE + 255) / 256), dim3(256), 0, stream,
                       ei, et, cursor, srcs);
    hipLaunchKernelGGL(graph_bounds, dim3((NN + 256) / 256), dim3(256), 0, stream,
                       batch, gstart);

    // ---------- layer 1 ----------
    hipLaunchKernelGGL(zero_f32, dim3(8192), dim3(256), 0, stream,
                       (float*)Af, (long long)NT * KT1 * 256);   // pads must be zero
    hipLaunchKernelGGL(segment_mean_frag, dim3(((size_t)NBINS * 32 + 255) / 256), dim3(256), 0, stream,
                       offs, srcs, x, Af, KT1, SEGW1, IN_DIM);
    hipLaunchKernelGGL(feat_to_frag, dim3(((long long)NN * IN_DIM + 255) / 256), dim3(256), 0, stream,
                       x, Af, KT1, SEGW1, IN_DIM);
    hipLaunchKernelGGL(gemm_frag, dim3((NT + 7) / 8), dim3(256), 0, stream,
                       Af, B1f, b1, h1, KT1, 1);

    // ---------- layer 2 ----------
    hipLaunchKernelGGL(segment_mean_frag, dim3(((size_t)NBINS * 32 + 255) / 256), dim3(256), 0, stream,
                       offs, srcs, h1, Af, KT2, SEGW2, HID);
    hipLaunchKernelGGL(feat_to_frag, dim3(((long long)NN * HID + 255) / 256), dim3(256), 0, stream,
                       h1, Af, KT2, SEGW2, HID);
    hipLaunchKernelGGL(gemm_frag, dim3((NT + 7) / 8), dim3(256), 0, stream,
                       Af, B2f, b2, h2, KT2, 1);

    // ---------- pool + head ----------
    hipLaunchKernelGGL(pool_mean, dim3((GG * 32) / 256), dim3(256), 0, stream,
                       h2, gstart, pooled);
    hipLaunchKernelGGL(head, dim3((GG * NCLS + 255) / 256), dim3(256), 0, stream,
                       pooled, lw, lb, out);
}
